// EgnnCleavageModel_24472723653186
// MI455X (gfx1250) — matrix operand order repaired and checked by reading, not compiled, not run on hardware
//
#include <hip/hip_runtime.h>
#include <hip/hip_bf16.h>
#include <cstddef>

typedef __attribute__((ext_vector_type(16))) _Float16 v16h;
typedef __attribute__((ext_vector_type(8)))  _Float16 v8h;
typedef __attribute__((ext_vector_type(8)))  float    v8f;
typedef __attribute__((ext_vector_type(4)))  float    f4;

#define NB    4
#define NN    384
#define ND    128
#define NH    64
#define NM    32
#define NV    1357
#define NROWS (NB * NN)   // 1536

__device__ __forceinline__ float silu_f(float x) {
    return x / (1.0f + __expf(-x));
}

__device__ __forceinline__ v8f wmma_f16(v16h a, v16h b, v8f c) {
    return __builtin_amdgcn_wmma_f32_16x16x32_f16(false, a, false, b, (short)0, c, false, false);
}

// ---------------------------------------------------------------------------
// proj_kernel: Pi = feats @ W1a + b1 ; Pj = feats @ W1b
// one wave per 16 node-rows. W1a = edge_w1 rows [0,128), W1b = rows [128,256).
// ---------------------------------------------------------------------------
__global__ __launch_bounds__(32) void proj_kernel(
    const float* __restrict__ f_in,   // [1536,128]
    const float* __restrict__ ew1,    // layer base: [257,64]
    const float* __restrict__ eb1,    // [64]
    float* __restrict__ Pi,           // [1536,64]
    float* __restrict__ Pj)           // [1536,64]
{
    const int lane = threadIdx.x;
    const int n0 = lane & 15;
    const int h  = lane >> 4;
    const int r0 = blockIdx.x * 16;
    const int row = r0 + n0;

    // A fragments: 16 rows x 128 K, f32 -> f16, vectorized b128 loads
    const f4* frow4 = (const f4*)(f_in + (size_t)row * ND);
    v16h a[4];
#pragma unroll
    for (int kc = 0; kc < 4; ++kc) {
        f4 u0 = frow4[kc * 8 + 2 * h];
        f4 u1 = frow4[kc * 8 + 2 * h + 1];
        f4 u2 = frow4[kc * 8 + 4 + 2 * h];
        f4 u3 = frow4[kc * 8 + 5 + 2 * h];
#pragma unroll
        for (int q = 0; q < 4; ++q) {
            a[kc][q]      = (_Float16)u0[q];
            a[kc][4 + q]  = (_Float16)u1[q];
            a[kc][8 + q]  = (_Float16)u2[q];
            a[kc][12 + q] = (_Float16)u3[q];
        }
    }

#pragma unroll
    for (int side = 0; side < 2; ++side) {
        const float* W = ew1 + (size_t)side * 128 * 64;
        float* Out = side ? Pj : Pi;
#pragma unroll
        for (int nt = 0; nt < 4; ++nt) {
            float bias = side ? 0.0f : eb1[nt * 16 + n0];
            v8f c;
#pragma unroll
            for (int v = 0; v < 8; ++v) c[v] = bias;
#pragma unroll
            for (int kc = 0; kc < 4; ++kc) {
                v16h bf;
#pragma unroll
                for (int p = 0; p < 16; ++p)
                    bf[p] = (_Float16)W[(size_t)(kc * 32 + 8 * h + p + (p & 8)) * 64 + nt * 16 + n0];
                c = wmma_f16(a[kc], bf, c);
            }
#pragma unroll
            for (int v = 0; v < 8; ++v)
                Out[(size_t)(r0 + v + 8 * h) * 64 + nt * 16 + n0] = c[v];
        }
    }
}

// ---------------------------------------------------------------------------
// edge_kernel: one wave per (b,i). Fused edge MLP + coor MLP + reductions.
// ---------------------------------------------------------------------------
__global__ __launch_bounds__(32) void edge_kernel(
    const float* __restrict__ Pi,     // [1536,64] (b1 folded)
    const float* __restrict__ Pj,     // [1536,64]
    const float* __restrict__ c_in,   // [1536,3]
    const unsigned char* __restrict__ mask, // [1536]
    const float* __restrict__ w1c,    // [64]  (edge_w1 row 256)
    const float* __restrict__ ew2,    // [64,32]
    const float* __restrict__ eb2,    // [32]
    const float* __restrict__ cw1,    // [32,64]
    const float* __restrict__ cb1,    // [64]
    const float* __restrict__ cw2v,   // [64]
    const float* __restrict__ cb2p,   // [1]
    float* __restrict__ mi_out,       // [1536,32]
    float* __restrict__ c_out)        // [1536,3]
{
    __shared__ __align__(16) float s_pi[64];
    __shared__ __align__(16) float s_w1c[64];
    __shared__ __align__(16) _Float16 s_m[16 * 32];

    const int lane = threadIdx.x;
    const int n0 = lane & 15;
    const int h  = lane >> 4;
    const int bi = blockIdx.x;
    const int b  = bi / NN;

    // stage Pi row (lane-half 0) and w1c (lane-half 1) via b128 each
    {
        const f4* src = h ? (const f4*)w1c : (const f4*)(Pi + (size_t)bi * 64);
        f4* dst = h ? (f4*)s_w1c : (f4*)s_pi;
        dst[n0] = src[n0];
    }
    __syncthreads();

    // valid-neighbor count for this batch
    float cnt = 0.0f;
    for (int t = lane; t < NN; t += 32) cnt += mask[b * NN + t] ? 1.0f : 0.0f;
#pragma unroll
    for (int m = 1; m < 32; m <<= 1) cnt += __shfl_xor(cnt, m, 32);
    const float mfi = mask[bi] ? 1.0f : 0.0f;
    const float nvalid = fmaxf(mfi * cnt, 1.0f);

    const float cix = c_in[bi * 3 + 0];
    const float ciy = c_in[bi * 3 + 1];
    const float ciz = c_in[bi * 3 + 2];

    // resident weight fragments (cold path: scalar loads, once per wave)
    v16h be[2][2];
#pragma unroll
    for (int kc = 0; kc < 2; ++kc)
#pragma unroll
        for (int nt = 0; nt < 2; ++nt)
#pragma unroll
            for (int p = 0; p < 16; ++p)
                be[kc][nt][p] = (_Float16)ew2[(size_t)(kc * 32 + 8 * h + p + (p & 8)) * 32 + nt * 16 + n0];
    v16h bc[4];
#pragma unroll
    for (int nt = 0; nt < 4; ++nt)
#pragma unroll
        for (int p = 0; p < 16; ++p)
            bc[nt][p] = (_Float16)cw1[(size_t)(8 * h + p + (p & 8)) * 64 + nt * 16 + n0];

    const float b2n0 = eb2[n0];
    const float b2n1 = eb2[n0 + 16];
    float cb1n[4], cw2n[4];
#pragma unroll
    for (int nt = 0; nt < 4; ++nt) {
        cb1n[nt] = cb1[nt * 16 + n0];
        cw2n[nt] = cw2v[nt * 16 + n0];
    }
    const float cb2s = cb2p[0];

    // cached f4 views of the per-wave constants
    const f4* pi4 = (const f4*)s_pi;
    const f4* wc4 = (const f4*)s_w1c;

    float macc0 = 0.0f, macc1 = 0.0f;
    float dax = 0.0f, day = 0.0f, daz = 0.0f;

    // run r (0..7) -> f4 index within a 64-float row; runs 0..3 feed a0, 4..7 feed a1
    const int off4[4] = {0, 1, 4, 5};

    for (int jb = 0; jb < NN; jb += 16) {
        if (jb + 16 < NN)  // stream next Pj tile toward L1
            __builtin_prefetch(Pj + (size_t)(b * NN + jb + 16 + n0) * 64, 0, 0);

        // squared distance for my A-row (row r = n0)
        const int jr = b * NN + jb + n0;
        const float dx = cix - c_in[jr * 3 + 0];
        const float dy = ciy - c_in[jr * 3 + 1];
        const float dz = ciz - c_in[jr * 3 + 2];
        const float dist = dx * dx + dy * dy + dz * dz;
        const f4* pj4 = (const f4*)(Pj + (size_t)jr * 64);

        // A = silu(Pi + Pj + dist*w1c + b1), built directly in WMMA A layout
        v16h a0, a1;
#pragma unroll
        for (int r = 0; r < 8; ++r) {
            const int fi = off4[r & 3] + 2 * h + ((r >= 4) ? 8 : 0);
            f4 P  = pj4[fi];
            f4 PI = pi4[fi];
            f4 W  = wc4[fi];
#pragma unroll
            for (int q = 0; q < 4; ++q) {
                _Float16 hv = (_Float16)silu_f(PI[q] + P[q] + dist * W[q]);
                if (r < 4) a0[(r & 3) * 4 + q] = hv;
                else       a1[(r & 3) * 4 + q] = hv;
            }
        }

        // m_tile = [16j x 32] = A @ edge_w2 + b2
        v8f m0, m1;
#pragma unroll
        for (int v = 0; v < 8; ++v) { m0[v] = b2n0; m1[v] = b2n1; }
        m0 = wmma_f16(a0, be[0][0], m0);
        m0 = wmma_f16(a1, be[1][0], m0);
        m1 = wmma_f16(a0, be[0][1], m1);
        m1 = wmma_f16(a1, be[1][1], m1);

        // mask, silu, m_i accumulation, LDS relayout for coor GEMM
        float mfrow[8];
#pragma unroll
        for (int v = 0; v < 8; ++v)
            mfrow[v] = (mask[b * NN + jb + v + 8 * h] ? 1.0f : 0.0f) * mfi;
#pragma unroll
        for (int v = 0; v < 8; ++v) {
            float mm0 = silu_f(m0[v]) * mfrow[v];
            float mm1 = silu_f(m1[v]) * mfrow[v];
            macc0 += mm0;
            macc1 += mm1;
            s_m[(v + 8 * h) * 32 + n0]      = (_Float16)mm0;
            s_m[(v + 8 * h) * 32 + n0 + 16] = (_Float16)mm1;
        }
        __syncthreads();
        v16h am;
        {
            v8h t0 = *(const v8h*)&s_m[n0 * 32 + 8 * h];
            v8h t1 = *(const v8h*)&s_m[n0 * 32 + 16 + 8 * h];
#pragma unroll
            for (int p = 0; p < 8; ++p) { am[p] = t0[p]; am[p + 8] = t1[p]; }
        }
        __syncthreads();

        // coor MLP: cw_j = silu(m @ cw1 + cb1) . cw2 + cb2
        float part[8];
#pragma unroll
        for (int v = 0; v < 8; ++v) part[v] = 0.0f;
#pragma unroll
        for (int nt = 0; nt < 4; ++nt) {
            v8f ch;
#pragma unroll
            for (int v = 0; v < 8; ++v) ch[v] = cb1n[nt];
            ch = wmma_f16(am, bc[nt], ch);
#pragma unroll
            for (int v = 0; v < 8; ++v) part[v] += silu_f(ch[v]) * cw2n[nt];
        }
#pragma unroll
        for (int v = 0; v < 8; ++v) {
#pragma unroll
            for (int m = 1; m < 16; m <<= 1) part[v] += __shfl_xor(part[v], m, 32);
        }
#pragma unroll
        for (int v = 0; v < 8; ++v) {
            float cw = (part[v] + cb2s) * mfrow[v];
            int j = b * NN + jb + v + 8 * h;
            dax += (cix - c_in[j * 3 + 0]) * cw;
            day += (ciy - c_in[j * 3 + 1]) * cw;
            daz += (ciz - c_in[j * 3 + 2]) * cw;
        }
    }

    macc0 += __shfl_xor(macc0, 16, 32);
    macc1 += __shfl_xor(macc1, 16, 32);
    dax += __shfl_xor(dax, 16, 32);
    day += __shfl_xor(day, 16, 32);
    daz += __shfl_xor(daz, 16, 32);

    if (lane < 16) {
        mi_out[(size_t)bi * 32 + n0]      = macc0;
        mi_out[(size_t)bi * 32 + n0 + 16] = macc1;
    }
    if (lane == 0) {
        float inv = 1.0f / nvalid;
        c_out[bi * 3 + 0] = cix + dax * inv;
        c_out[bi * 3 + 1] = ciy + day * inv;
        c_out[bi * 3 + 2] = ciz + daz * inv;
    }
}

// ---------------------------------------------------------------------------
// node_kernel: feats' = (feats + silu([feats|m_i]@W1 + b1)@W2 + b2) * mf
// ---------------------------------------------------------------------------
__global__ __launch_bounds__(32) void node_kernel(
    const float* __restrict__ f_in,   // [1536,128]
    const float* __restrict__ mi,     // [1536,32]
    const float* __restrict__ w1,     // [160,64]
    const float* __restrict__ b1,     // [64]
    const float* __restrict__ w2,     // [64,128]
    const float* __restrict__ b2,     // [128]
    const unsigned char* __restrict__ mask,
    float* __restrict__ f_out)        // [1536,128]
{
    __shared__ __align__(16) _Float16 s_h[16 * 64];
    const int lane = threadIdx.x;
    const int n0 = lane & 15;
    const int h  = lane >> 4;
    const int r0 = blockIdx.x * 16;
    const int row = r0 + n0;

    // A1 = [feats | m_i], K = 160 (kc 0..3 = feats, kc 4 = m_i), b128 loads
    const f4* frow4 = (const f4*)(f_in + (size_t)row * ND);
    const f4* mrow4 = (const f4*)(mi + (size_t)row * NM);
    v16h a[5];
#pragma unroll
    for (int kc = 0; kc < 5; ++kc) {
        f4 u0, u1, u2, u3;
        if (kc < 4) {
            u0 = frow4[kc * 8 + 2 * h];
            u1 = frow4[kc * 8 + 2 * h + 1];
            u2 = frow4[kc * 8 + 4 + 2 * h];
            u3 = frow4[kc * 8 + 5 + 2 * h];
        } else {
            u0 = mrow4[2 * h];
            u1 = mrow4[2 * h + 1];
            u2 = mrow4[4 + 2 * h];
            u3 = mrow4[5 + 2 * h];
        }
#pragma unroll
        for (int q = 0; q < 4; ++q) {
            a[kc][q]      = (_Float16)u0[q];
            a[kc][4 + q]  = (_Float16)u1[q];
            a[kc][8 + q]  = (_Float16)u2[q];
            a[kc][12 + q] = (_Float16)u3[q];
        }
    }

    // GEMM1 -> h[16x64], silu, stash in LDS in row-major f16
#pragma unroll
    for (int nt = 0; nt < 4; ++nt) {
        v8f c;
        float bias = b1[nt * 16 + n0];
#pragma unroll
        for (int v = 0; v < 8; ++v) c[v] = bias;
#pragma unroll
        for (int kc = 0; kc < 5; ++kc) {
            v16h bf;
#pragma unroll
            for (int p = 0; p < 16; ++p)
                bf[p] = (_Float16)w1[(size_t)(kc * 32 + 8 * h + p + (p & 8)) * 64 + nt * 16 + n0];
            c = wmma_f16(a[kc], bf, c);
        }
#pragma unroll
        for (int v = 0; v < 8; ++v)
            s_h[(v + 8 * h) * 64 + nt * 16 + n0] = (_Float16)silu_f(c[v]);
    }
    __syncthreads();

    // A2 from LDS, K = 64, 16B ds loads
    v16h a2[2];
#pragma unroll
    for (int kc = 0; kc < 2; ++kc) {
        v8h t0 = *(const v8h*)&s_h[n0 * 64 + kc * 32 + 8 * h];
        v8h t1 = *(const v8h*)&s_h[n0 * 64 + kc * 32 + 16 + 8 * h];
#pragma unroll
        for (int p = 0; p < 8; ++p) { a2[kc][p] = t0[p]; a2[kc][p + 8] = t1[p]; }
    }

    // GEMM2 -> [16x128], residual + mask
#pragma unroll
    for (int nt = 0; nt < 8; ++nt) {
        v8f c;
        float bias = b2[nt * 16 + n0];
#pragma unroll
        for (int v = 0; v < 8; ++v) c[v] = bias;
#pragma unroll
        for (int kc = 0; kc < 2; ++kc) {
            v16h bf;
#pragma unroll
            for (int p = 0; p < 16; ++p)
                bf[p] = (_Float16)w2[(size_t)(kc * 32 + 8 * h + p + (p & 8)) * ND + nt * 16 + n0];
            c = wmma_f16(a2[kc], bf, c);
        }
#pragma unroll
        for (int v = 0; v < 8; ++v) {
            int gr = r0 + v + 8 * h;
            float mf = mask[gr] ? 1.0f : 0.0f;
            f_out[(size_t)gr * ND + nt * 16 + n0] =
                (f_in[(size_t)gr * ND + nt * 16 + n0] + c[v]) * mf;
        }
    }
}

// ---------------------------------------------------------------------------
// head: masked mean pool + linear
// ---------------------------------------------------------------------------
__global__ __launch_bounds__(128) void pool_kernel(
    const float* __restrict__ f, const unsigned char* __restrict__ mask,
    float* __restrict__ pooled)
{
    const int b = blockIdx.x;
    const int d = threadIdx.x;
    float acc = 0.0f, den = 0.0f;
    for (int n = 0; n < NN; ++n) {
        float mf = mask[b * NN + n] ? 1.0f : 0.0f;
        acc += f[((size_t)b * NN + n) * ND + d] * mf;
        den += mf;
    }
    pooled[b * ND + d] = acc / fmaxf(den, 1.0f);
}

__global__ __launch_bounds__(128) void head_kernel(
    const float* __restrict__ pooled, const float* __restrict__ hw,
    const float* __restrict__ hb, float* __restrict__ out)
{
    const int idx = blockIdx.x * blockDim.x + threadIdx.x;
    if (idx >= NB * NV) return;
    const int b = idx / NV;
    const int v = idx % NV;
    float acc = hb[v];
#pragma unroll 4
    for (int d = 0; d < ND; ++d)
        acc += pooled[b * ND + d] * hw[(size_t)d * NV + v];
    out[(size_t)b * NV + v] = acc;
}

// ---------------------------------------------------------------------------
extern "C" void kernel_launch(void* const* d_in, const int* in_sizes, int n_in,
                              void* d_out, int out_size, void* d_ws, size_t ws_size,
                              hipStream_t stream) {
    const float* feats   = (const float*)d_in[0];
    const float* coors   = (const float*)d_in[1];
    const unsigned char* mask = (const unsigned char*)d_in[2];
    const float* edge_w1 = (const float*)d_in[3];
    const float* edge_b1 = (const float*)d_in[4];
    const float* edge_w2 = (const float*)d_in[5];
    const float* edge_b2 = (const float*)d_in[6];
    const float* coor_w1 = (const float*)d_in[7];
    const float* coor_b1 = (const float*)d_in[8];
    const float* coor_w2 = (const float*)d_in[9];
    const float* coor_b2 = (const float*)d_in[10];
    const float* node_w1 = (const float*)d_in[11];
    const float* node_b1 = (const float*)d_in[12];
    const float* node_w2 = (const float*)d_in[13];
    const float* node_b2 = (const float*)d_in[14];
    const float* head_w  = (const float*)d_in[15];
    const float* head_b  = (const float*)d_in[16];

    char* w = (char*)d_ws;
    size_t off = 0;
    float* Pi     = (float*)(w + off); off += (size_t)NROWS * 64 * 4;
    float* Pjb    = (float*)(w + off); off += (size_t)NROWS * 64 * 4;
    float* Mi     = (float*)(w + off); off += (size_t)NROWS * NM * 4;
    float* featsX = (float*)(w + off); off += (size_t)NROWS * ND * 4;
    float* featsY = (float*)(w + off); off += (size_t)NROWS * ND * 4;
    float* coorsX = (float*)(w + off); off += (size_t)NROWS * 3 * 4;
    float* coorsY = (float*)(w + off); off += (size_t)NROWS * 3 * 4;
    float* pooled = (float*)(w + off); off += (size_t)NB * ND * 4;

    for (int l = 0; l < 2; ++l) {
        const float* f_in  = (l == 0) ? feats : featsX;
        const float* c_in  = (l == 0) ? coors : coorsX;
        float* f_out = (l == 0) ? featsX : featsY;
        float* c_out = (l == 0) ? coorsX : coorsY;

        proj_kernel<<<NROWS / 16, 32, 0, stream>>>(
            f_in, edge_w1 + (size_t)l * 257 * 64, edge_b1 + l * 64, Pi, Pjb);

        edge_kernel<<<NROWS, 32, 0, stream>>>(
            Pi, Pjb, c_in, mask,
            edge_w1 + (size_t)l * 257 * 64 + 256 * 64,
            edge_w2 + (size_t)l * 64 * 32, edge_b2 + l * 32,
            coor_w1 + (size_t)l * 32 * 64, coor_b1 + l * 64,
            coor_w2 + (size_t)l * 64, coor_b2 + l,
            Mi, c_out);

        node_kernel<<<NROWS / 16, 32, 0, stream>>>(
            f_in, Mi,
            node_w1 + (size_t)l * 160 * 64, node_b1 + l * 64,
            node_w2 + (size_t)l * 64 * 128, node_b2 + l * 128,
            mask, f_out);
    }

    pool_kernel<<<NB, 128, 0, stream>>>(featsY, mask, pooled);
    head_kernel<<<(NB * NV + 127) / 128, 128, 0, stream>>>(pooled, head_w, head_b, (float*)d_out);
}
